// Head_60859686584498
// MI455X (gfx1250) — matrix operand order repaired and checked by reading, not compiled, not run on hardware
//
#include <hip/hip_runtime.h>

// ---------------------------------------------------------------------------
// MI455X (gfx1250, wave32) flash-attention: bf16 WMMA + async-to-LDS staging
// ---------------------------------------------------------------------------

typedef __attribute__((ext_vector_type(16))) __bf16 v16bf;
typedef __attribute__((ext_vector_type(8)))  float  v8f;

struct alignas(16) U128 { unsigned x, y, z, w; };
union BFTile { v16bf v; U128 q[2]; };             // 32 bytes = one WMMA A/B operand
union Pack8  { U128 q; unsigned short h[8]; };    // 8 bf16 packed store

constexpr int BB = 4;
constexpr int TT = 4096;
constexpr int CC = 1024;
constexpr int HH = 64;

__device__ __forceinline__ unsigned short f2bf(float f) {
  union { float f; unsigned u; } c; c.f = f;
  unsigned u = c.u;
  return (unsigned short)((u + 0x7FFFu + ((u >> 16) & 1u)) >> 16);  // RNE
}

__device__ __forceinline__ v8f wmma_bf16(v16bf a, v16bf b, v8f c) {
  // D = A(16x32 bf16) * B(32x16 bf16) + C(16x16 f32)
  return __builtin_amdgcn_wmma_f32_16x16x32_bf16(
      /*neg_a=*/false, a, /*neg_b=*/false, b,
      /*c_mod=*/(short)0, c, /*reuse_a=*/false, /*reuse_b=*/false);
}

// 32-bit LDS byte address of a generic pointer into __shared__
__device__ __forceinline__ unsigned lds_off_of(const unsigned short* p) {
  return (unsigned)(unsigned long long)
         (__attribute__((address_space(3))) const unsigned short*)p;
}

// One async 16-byte global->LDS transfer per lane (tracked with ASYNCcnt)
__device__ __forceinline__ void async_cp16(unsigned lds_off, const void* g) {
  unsigned long long ga = (unsigned long long)(size_t)g;
  asm volatile("global_load_async_to_lds_b128 %0, %1, off"
               :: "v"(lds_off), "v"(ga) : "memory");
}

// ----------------------------- kernel 0: f32 -> bf16 -----------------------
__global__ void cvt_bf16_kernel(const float* __restrict__ in,
                                unsigned short* __restrict__ out,
                                int n2, float scale) {
  int i = blockIdx.x * blockDim.x + threadIdx.x;
  if (i < n2) {
    float a = in[2 * i + 0] * scale;
    float b = in[2 * i + 1] * scale;
    ((unsigned*)out)[i] = (unsigned)f2bf(a) | ((unsigned)f2bf(b) << 16);
  }
}

// ----------------------- kernel 1: q/k/v projections -----------------------
// One wave computes one 16x16 output tile: rows = 16 tokens, cols = 16 of 64 h.
// colTile 0..3 -> q (1/sqrt(d) folded into Wq), 4..7 -> k, 8..11 -> v.
__global__ void __launch_bounds__(256)
proj16_kernel(const unsigned short* __restrict__ xb,
              const unsigned short* __restrict__ wq,
              const unsigned short* __restrict__ wk,
              const unsigned short* __restrict__ wv,
              unsigned short* __restrict__ qb,
              unsigned short* __restrict__ kb,
              unsigned short* __restrict__ vT) {
  const int wave = (blockIdx.x * blockDim.x + threadIdx.x) >> 5;
  const int lane = threadIdx.x & 31;
  const int rowTile = wave / 12;
  const int colTile = wave % 12;
  const int proj = colTile >> 2;
  const int h0 = (colTile & 3) << 4;
  const unsigned short* w = (proj == 0) ? wq : (proj == 1 ? wk : wv);

  const int row = lane & 15;        // A-layout: lane owns matrix row (= token)
  const int g   = lane >> 4;        // half-wave selector for K offsets
  const int col = row;              // B/D-layout: lane owns matrix column
  const int t0  = rowTile << 4;

  const unsigned short* aBase = xb + (size_t)(t0 + row) * CC + 16 * g;
  const unsigned short* bBase = w + (size_t)(h0 + col) * CC + 16 * g;

  v8f acc = {};
  for (int k0 = 0; k0 < CC; k0 += 32) {
    BFTile a, bt;
    a.q[0]  = *(const U128*)(aBase + k0);
    a.q[1]  = *(const U128*)(aBase + k0 + 8);
    bt.q[0] = *(const U128*)(bBase + k0);
    bt.q[1] = *(const U128*)(bBase + k0 + 8);
    acc = wmma_bf16(a.v, bt.v, acc);
  }

  if (proj < 2) {
    unsigned short* dst = (proj == 0) ? qb : kb;
#pragma unroll
    for (int r = 0; r < 8; r++)
      dst[(size_t)(t0 + r + 8 * g) * HH + h0 + col] = f2bf(acc[r]);
  } else {
    // v stored transposed [B][64][T] bf16 so PV B-loads are contiguous in s
    const int b  = t0 / TT;
    const int tl = (t0 % TT) + 8 * g;
    Pack8 p;
#pragma unroll
    for (int r = 0; r < 8; r++) p.h[r] = f2bf(acc[r]);
    *(U128*)(vT + ((size_t)b * HH + h0 + col) * TT + tl) = p.q;
  }
}

// ----------------------- kernel 2: flash attention -------------------------
// 4 waves per workgroup, each wave owns 16 query rows of the same batch.
// K/V tiles (32 keys) are async-staged into double-buffered LDS once per WG
// and shared by all 4 waves (4x less L2 traffic than per-wave global reads).
__global__ void __launch_bounds__(128)
attn_kernel(const unsigned short* __restrict__ qb,
            const unsigned short* __restrict__ kb,
            const unsigned short* __restrict__ vT,
            const int* __restrict__ pm,
            float* __restrict__ out) {
  constexpr int WAVES = 4;
  constexpr int NIT = TT / 32;                             // 128 key tiles
  __shared__ alignas(16) unsigned short sK[2][32 * HH];    // 2 x 4 KB, [s][h]
  __shared__ alignas(16) unsigned short sV[2][HH * 32];    // 2 x 4 KB, [h][s]
  __shared__ alignas(16) unsigned short sP[WAVES][16 * 32];// P transpose staging

  const int tid  = threadIdx.x;
  const int wl   = tid >> 5;
  const int lane = tid & 31;
  const int qt   = blockIdx.x * WAVES + wl;       // query tile id, 0..1023
  const int b    = qt >> 8;                       // 256 tiles per batch; all
  const int t0   = (qt & 255) << 4;               //   waves of a WG share b
  const int row  = lane & 15;
  const int g    = lane >> 4;
  const int col  = row;
  unsigned short* myP = sP[wl];
  const float NEG_INF = -__builtin_inff();

  const int* pmB = pm + b * TT;
  const unsigned short* kB = kb + (size_t)b * TT * HH;
  const unsigned short* vB = vT + (size_t)b * HH * TT;

  // Q tile as two A operands (K = h 0..31 and 32..63); 1/sqrt(d) pre-folded.
  const unsigned short* qrow = qb + (size_t)(b * TT + t0 + row) * HH;
  BFTile aq0, aq1;
  aq0.q[0] = *(const U128*)(qrow + 16 * g);
  aq0.q[1] = *(const U128*)(qrow + 8 + 16 * g);
  aq1.q[0] = *(const U128*)(qrow + 32 + 16 * g);
  aq1.q[1] = *(const U128*)(qrow + 40 + 16 * g);

  // Cooperative async staging of one 32-key tile: k tile is a contiguous 4 KB
  // region (rows s0..s0+31 of [T][64]); v tile is 64 rows of 64 B from [64][T].
  // 128 threads x 4 chunks of 16 B = 8 KB; 4 async issues per wave.
  auto issue_tile = [&](int s0, int buf) {
#pragma unroll
    for (int j = 0; j < 2; j++) {
      const int c = tid + j * 128;                     // chunk id, 0..255
      async_cp16(lds_off_of(&sK[buf][c * 8]), kB + (size_t)s0 * HH + c * 8);
      const int h = c >> 2, sub = c & 3;
      async_cp16(lds_off_of(&sV[buf][h * 32 + sub * 8]),
                 vB + (size_t)h * TT + s0 + sub * 8);
    }
  };

  v8f z = {};
  v8f acc[4] = {z, z, z, z};                      // 16 x 64 f32 output tile
  float m[8], l[8];
#pragma unroll
  for (int r = 0; r < 8; r++) { m[r] = NEG_INF; l[r] = 0.0f; }

  issue_tile(0, 0);

  for (int it = 0; it < NIT; ++it) {
    const int s0 = it * 32;
    const int cur = it & 1;

    if (it + 1 < NIT) {
      issue_tile(s0 + 32, cur ^ 1);                // buf^1 sealed by last barrier
      // async loads complete in order: <=4 outstanding => tile `it` has landed
      asm volatile("s_wait_asynccnt 0x4" ::: "memory");
    } else {
      asm volatile("s_wait_asynccnt 0x0" ::: "memory");
    }
    __syncthreads();                               // whole tile visible to WG

    // ---- scores S = Q K^T (16 x 32): two 16x16 D tiles, K=64 split 2x32 ----
    v8f sc[2];
#pragma unroll
    for (int nh = 0; nh < 2; nh++) {
      const unsigned short* krow = &sK[cur][(nh * 16 + col) * HH + 16 * g];
      BFTile b0, b1;
      b0.q[0] = *(const U128*)(krow);
      b0.q[1] = *(const U128*)(krow + 8);
      b1.q[0] = *(const U128*)(krow + 32);
      b1.q[1] = *(const U128*)(krow + 40);
      v8f c = {};
      c = wmma_bf16(aq0.v, b0.v, c);
      c = wmma_bf16(aq1.v, b1.v, c);
      sc[nh] = c;
    }

    // ---- key-padding mask (pre-softmax) + per-row tile max ----
    float tmax[8];
#pragma unroll
    for (int r = 0; r < 8; r++) tmax[r] = NEG_INF;
#pragma unroll
    for (int nh = 0; nh < 2; nh++) {
      const bool dead = (pmB[s0 + nh * 16 + col] == 0);
#pragma unroll
      for (int r = 0; r < 8; r++) {
        float sv = dead ? NEG_INF : sc[nh][r];
        sc[nh][r] = sv;
        tmax[r] = fmaxf(tmax[r], sv);
      }
    }

    // ---- online softmax row stats (xor-shuffles stay in 16-lane half) ----
    float scl[8], mu[8];
#pragma unroll
    for (int r = 0; r < 8; r++) {
      float tm = tmax[r];
#pragma unroll
      for (int xm = 1; xm < 16; xm <<= 1) tm = fmaxf(tm, __shfl_xor(tm, xm, 32));
      float mn   = fmaxf(m[r], tm);
      float muse = (mn == NEG_INF) ? 0.0f : mn;   // fully-masked-so-far guard
      scl[r] = __expf(m[r] - muse);               // -> 0 when m was -inf
      mu[r]  = muse;
      m[r]   = mn;
    }

    // ---- P = exp(S - m), row sums, D->A transpose via per-wave LDS tile ----
    float rsum[8] = {0, 0, 0, 0, 0, 0, 0, 0};
#pragma unroll
    for (int nh = 0; nh < 2; nh++) {
#pragma unroll
      for (int r = 0; r < 8; r++) {
        float p = __expf(sc[nh][r] - mu[r]);      // exp(-inf) = 0 for masked cols
        rsum[r] += p;
        myP[(r + 8 * g) * 32 + nh * 16 + col] = f2bf(p);
      }
    }
#pragma unroll
    for (int r = 0; r < 8; r++) {
      float rs = rsum[r];
#pragma unroll
      for (int xm = 1; xm < 16; xm <<= 1) rs += __shfl_xor(rs, xm, 32);
      l[r] = l[r] * scl[r] + rs;
#pragma unroll
      for (int h = 0; h < 4; h++) acc[h][r] *= scl[r];
    }

    // P tile in A-layout (16 rows x 32 keys); DS ops are in-order per wave.
    BFTile pa;
    const unsigned short* prow = myP + row * 32 + 16 * g;
    pa.q[0] = *(const U128*)(prow);
    pa.q[1] = *(const U128*)(prow + 8);

    // ---- acc += P V  (K = 32 keys, 4 x 16 h columns) ----
#pragma unroll
    for (int h = 0; h < 4; h++) {
      const unsigned short* vrow = &sV[cur][(h * 16 + col) * 32 + 16 * g];
      BFTile bv;
      bv.q[0] = *(const U128*)(vrow);
      bv.q[1] = *(const U128*)(vrow + 8);
      acc[h] = wmma_bf16(pa.v, bv.v, acc[h]);
    }

    __syncthreads();   // seal buf[cur] before it is re-targeted next iteration
  }

  // ---- epilogue: 1/l, post-softmax query-row mask, f32 store ----
#pragma unroll
  for (int r = 0; r < 8; r++) {
    const int t = t0 + r + 8 * g;
    const float rl = (l[r] > 0.0f) ? (1.0f / l[r]) : 0.0f;
    const float rm = (pmB[t] == 0) ? 0.0f : rl;
#pragma unroll
    for (int h = 0; h < 4; h++)
      out[(size_t)(b * TT + t) * HH + h * 16 + col] = acc[h][r] * rm;
  }
}

// --------------------------------- launcher --------------------------------
extern "C" void kernel_launch(void* const* d_in, const int* in_sizes, int n_in,
                              void* d_out, int out_size, void* d_ws, size_t ws_size,
                              hipStream_t stream) {
  (void)in_sizes; (void)n_in; (void)out_size; (void)ws_size;

  const float* x  = (const float*)d_in[0];
  const int*   pm = (const int*)d_in[1];
  const float* Wk = (const float*)d_in[2];
  const float* Wq = (const float*)d_in[3];
  const float* Wv = (const float*)d_in[4];
  float* out = (float*)d_out;

  // workspace layout (bytes); total ~40.2 MB
  char* ws = (char*)d_ws;
  const size_t nX = (size_t)BB * TT * CC;     // 16,777,216
  const size_t nW = (size_t)HH * CC;          // 65,536
  const size_t nP = (size_t)BB * TT * HH;     // 1,048,576
  unsigned short* xb  = (unsigned short*)(ws);
  unsigned short* wqb = (unsigned short*)(ws + 2 * nX);
  unsigned short* wkb = (unsigned short*)(ws + 2 * (nX + nW));
  unsigned short* wvb = (unsigned short*)(ws + 2 * (nX + 2 * nW));
  unsigned short* qb  = (unsigned short*)(ws + 2 * (nX + 3 * nW));
  unsigned short* kb  = (unsigned short*)(ws + 2 * (nX + 3 * nW + nP));
  unsigned short* vT  = (unsigned short*)(ws + 2 * (nX + 3 * nW + 2 * nP));

  const float inv_sqrt_d = 0.125f;  // 1/sqrt(64), folded into Wq

  // 0) f32 -> bf16 conversions
  {
    int n2 = (int)(nX / 2);
    cvt_bf16_kernel<<<(n2 + 255) / 256, 256, 0, stream>>>(x, xb, n2, 1.0f);
    n2 = (int)(nW / 2);
    cvt_bf16_kernel<<<(n2 + 255) / 256, 256, 0, stream>>>(Wq, wqb, n2, inv_sqrt_d);
    cvt_bf16_kernel<<<(n2 + 255) / 256, 256, 0, stream>>>(Wk, wkb, n2, 1.0f);
    cvt_bf16_kernel<<<(n2 + 255) / 256, 256, 0, stream>>>(Wv, wvb, n2, 1.0f);
  }

  // 1) projections: (B*T/16) row tiles x 12 col tiles = 12288 waves, 8/block
  proj16_kernel<<<1536, 256, 0, stream>>>(xb, wqb, wkb, wvb, qb, kb, vT);

  // 2) attention: 1024 query tiles, 4 waves/block, async double-buffered K/V
  attn_kernel<<<256, 128, 0, stream>>>(qb, kb, vT, pm, out);
}
